// DCNv3_pytorch_6880537608758
// MI455X (gfx1250) — matrix operand order repaired and checked by reading, not compiled, not run on hardware
//
#include <hip/hip_runtime.h>
#include <hip/hip_bf16.h>
#include <stdint.h>

// ---------------- problem constants (from reference) ----------------
#define TN_    8          // T*B
#define HDIM   64
#define WDIM   64
#define CDIM   128
#define MEDDIM 512
#define HWDIM  4096       // 64*64
#define MROWS  32768      // TN * H * W
#define GRPS   4
#define GC_    32
#define PTAPS  9

typedef __attribute__((ext_vector_type(16))) _Float16 v16h;
typedef __attribute__((ext_vector_type(8)))  _Float16 v8h;
typedef __attribute__((ext_vector_type(8)))  float    v8f;

__device__ __forceinline__ float spikef(float v) {
    // forward value of Q_IFNode quant surrogate: clip(round(x), 0, 4)
    return fminf(fmaxf(rintf(v), 0.0f), 4.0f);
}

// async memory->LDS copies, tracked by ASYNCcnt (CDNA5 GLOBAL_LOAD_ASYNC_TO_LDS_B128).
// GVS addressing: saddr(u64) + vaddr(i32) + inst_offset; the 24-bit inst_offset advances
// BOTH the global and the LDS address, so consecutive 16B chunks reuse the same operands.
#define ASYNC_B128(OFFSTR, LDS, VOFF, SBASE)                                   \
    asm volatile("global_load_async_to_lds_b128 %0, %1, %2" OFFSTR             \
                 :: "v"(LDS), "v"(VOFF), "s"(SBASE) : "memory")

__device__ __forceinline__ void wait_async0() {
    asm volatile("s_wait_asynccnt 0" ::: "memory");
}

// ---------------- elementwise spike -> fp16 ----------------
__global__ __launch_bounds__(256) void k_spike(const float* __restrict__ in,
                                               _Float16* __restrict__ out, int n) {
    int i = blockIdx.x * 256 + threadIdx.x;
    if (i < n) out[i] = (_Float16)spikef(in[i]);
}

// ---------------- weight convert + zero-pad: w[N][K] f32 -> out[Npad][K] f16 ----------------
__global__ __launch_bounds__(256) void k_wpad(const float* __restrict__ w,
                                              _Float16* __restrict__ out,
                                              int N, int Npad, int K) {
    int i = blockIdx.x * 256 + threadIdx.x;
    if (i < Npad * K) {
        int n = i / K;
        out[i] = (_Float16)((n < N) ? w[i] : 0.0f);
    }
}

// ---------------- WMMA GEMM: C[M,N] = A[M,K] * Bw[N][K]^T, fused BN/bias/spike epilogue ------
// Block tile 64x128; each wave owns a 16x64 strip = 4 accumulators sharing one A fragment.
// Async double-buffered LDS staging (ASYNCcnt), K tiled by 64 (8 WMMAs per wave per tile).
// MODE 0: y = acc*s+b              -> f32 row-major [M,N]
// MODE 1: y = spike(acc*s+b)       -> f16 row-major [M,N]
// MODE 2: y = (acc+cb)*s+b         -> f32 stored NCHW per-tn (faithful raw-reshape scramble)
// MODE 3: y = spike((acc+cb)*s+b)  -> f32 stored NCHW per-tn
template <int MODE, int KDIM>
__global__ __launch_bounds__(256) void k_gemm(const _Float16* __restrict__ A,
                                              const _Float16* __restrict__ Bw,  // [Npad][KDIM]
                                              const float* __restrict__ bns,
                                              const float* __restrict__ bnb,
                                              const float* __restrict__ cbias,
                                              float* __restrict__ outf,
                                              _Float16* __restrict__ outh,
                                              int N) {
    __shared__ alignas(32) _Float16 ldsA[2][64 * 64];   // [buf][Mtile=64][Ktile=64]   16 KB
    __shared__ alignas(32) _Float16 ldsB[2][128 * 64];  // [buf][Ntile=128][Ktile=64]  32 KB

    const int m0 = blockIdx.x * 64;
    const int n0 = blockIdx.y * 128;
    const int t = threadIdx.x;
    const int wave = t >> 5;           // 8 waves
    const int lane = t & 31;
    const int wm = wave >> 1;          // 0..3 -> 16-row slab
    const int wn = wave & 1;           // 0..1 -> 64-col slab (4 x 16)
    const int lanelo = lane & 15;
    const int laneHalf = lane >> 4;

    const uint32_t ldsA0 = (uint32_t)(uintptr_t)&ldsA[0][0];
    const uint32_t ldsB0 = (uint32_t)(uintptr_t)&ldsB[0][0];

    // per-thread staging coordinates
    const int rowA = t >> 2;               // 0..63
    const int chA  = (t & 3) << 4;         // 0,16,32,48 halves  (32B -> 2x b128)
    const int rowB = t >> 1;               // 0..127
    const int chB  = (t & 1) << 5;         // 0,32 halves        (64B -> 4x b128)
    const uint32_t voffA = (uint32_t)(((m0 + rowA) * KDIM + chA) * 2);
    const uint32_t voffB = (uint32_t)(((n0 + rowB) * KDIM + chB) * 2);

    auto stage = [&](int k0, int buf) {
        const uint64_t baseA = (uint64_t)(uintptr_t)A + (uint64_t)k0 * 2;
        const uint64_t baseB = (uint64_t)(uintptr_t)Bw + (uint64_t)k0 * 2;
        const uint32_t la = ldsA0 + (uint32_t)(((buf << 12) + rowA * 64 + chA) * 2);
        const uint32_t lb = ldsB0 + (uint32_t)(((buf << 13) + rowB * 64 + chB) * 2);
        ASYNC_B128("",            la, voffA, baseA);
        ASYNC_B128(" offset:16",  la, voffA, baseA);
        ASYNC_B128("",            lb, voffB, baseB);
        ASYNC_B128(" offset:16",  lb, voffB, baseB);
        ASYNC_B128(" offset:32",  lb, voffB, baseB);
        ASYNC_B128(" offset:48",  lb, voffB, baseB);
    };

    v8f acc0 = {}, acc1 = {}, acc2 = {}, acc3 = {};
    constexpr int NT = KDIM / 64;

    stage(0, 0);
    wait_async0();
    __syncthreads();

    #pragma unroll
    for (int it = 0; it < NT; ++it) {
        if (it + 1 < NT) stage((it + 1) * 64, (it + 1) & 1);
        const int buf = it & 1;
        #pragma unroll
        for (int ks = 0; ks < 64; ks += 32) {
            // A fragment, contiguous K order matching B:
            //   lanes 0-15 row M hold K {0..15}; lanes 16-31 hold K {16..31}
            const _Float16* ar = &ldsA[buf][(wm * 16 + lanelo) * 64 + ks + laneHalf * 16];
            v8h alo = *(const v8h*)(ar);
            v8h ahi = *(const v8h*)(ar + 8);
            v16h a = __builtin_shufflevector(alo, ahi, 0, 1, 2, 3, 4, 5, 6, 7,
                                             8, 9, 10, 11, 12, 13, 14, 15);
            // B fragments: lane = column, contiguous K range (laneHalf*16 .. +15)
            const _Float16* br = &ldsB[buf][(wn * 64 + lanelo) * 64 + ks + laneHalf * 16];
            v16h b0 = *(const v16h*)(br);
            v16h b1 = *(const v16h*)(br + 16 * 64);
            v16h b2 = *(const v16h*)(br + 32 * 64);
            v16h b3 = *(const v16h*)(br + 48 * 64);
            acc0 = __builtin_amdgcn_wmma_f32_16x16x32_f16(false, a, false, b0, (short)0, acc0, false, false);
            acc1 = __builtin_amdgcn_wmma_f32_16x16x32_f16(false, a, false, b1, (short)0, acc1, false, false);
            acc2 = __builtin_amdgcn_wmma_f32_16x16x32_f16(false, a, false, b2, (short)0, acc2, false, false);
            acc3 = __builtin_amdgcn_wmma_f32_16x16x32_f16(false, a, false, b3, (short)0, acc3, false, false);
        }
        if (it + 1 < NT) {
            wait_async0();      // this wave's async writes for next tile are in LDS
            __syncthreads();    // everyone's writes visible, prev buffer free
        }
    }

    // epilogue: C/D layout -> VGPR r: M = r + 8*laneHalf, N = lanelo
    v8f accs[4] = {acc0, acc1, acc2, acc3};
    #pragma unroll
    for (int j = 0; j < 4; ++j) {
        const int col = n0 + wn * 64 + j * 16 + lanelo;
        if (col < N) {
            const float s = bns[col];
            const float bb = bnb[col];
            const float cb = (MODE >= 2) ? cbias[col] : 0.0f;
            #pragma unroll
            for (int r = 0; r < 8; ++r) {
                const int gr = m0 + wm * 16 + laneHalf * 8 + r;
                float v = accs[j][r];
                if (MODE >= 2) v += cb;
                v = v * s + bb;
                if (MODE == 0) {
                    outf[(size_t)gr * N + col] = v;
                } else if (MODE == 1) {
                    outh[(size_t)gr * N + col] = (_Float16)spikef(v);
                } else {
                    // faithful raw reshape: consumer reads per-tn flat (h*W+w)*N + ch
                    // from an NCHW block -> producer stores NCHW [tn][ch][pos]
                    const int tn = gr >> 12;
                    const int pos = gr & 4095;
                    const size_t idx = ((size_t)tn * N + col) * HWDIM + pos;
                    outf[idx] = (MODE == 3) ? spikef(v) : v;
                }
            }
        }
    }
}

// ---------------- depthwise 3x3 + BN + spike, channel-last ----------------
__global__ __launch_bounds__(256) void k_dw(const _Float16* __restrict__ in,
                                            const float* __restrict__ w,
                                            const float* __restrict__ s,
                                            const float* __restrict__ b,
                                            _Float16* __restrict__ out, int Ch) {
    const size_t idx = (size_t)blockIdx.x * 256 + threadIdx.x;
    const size_t total = (size_t)TN_ * HWDIM * Ch;
    if (idx >= total) return;
    const int c = (int)(idx % Ch);
    const size_t sp = idx / Ch;
    const int ww = (int)(sp & 63);
    const int hh = (int)((sp >> 6) & 63);
    const int tn = (int)(sp >> 12);
    const float* wc = w + c * 9;
    float acc = 0.0f;
    #pragma unroll
    for (int dy = -1; dy <= 1; ++dy) {
        #pragma unroll
        for (int dx = -1; dx <= 1; ++dx) {
            const int y = hh + dy, x = ww + dx;
            if (y >= 0 && y < HDIM && x >= 0 && x < WDIM)
                acc += (float)in[(((size_t)tn * HDIM + y) * WDIM + x) * Ch + c] *
                       wc[(dy + 1) * 3 + (dx + 1)];
        }
    }
    out[idx] = (_Float16)spikef(acc * s[c] + b[c]);
}

// ---------------- DCNv3 core: one wave32 per (tn, hw, g); lane = channel in group ------------
// Fused output spike: store spike(acc) directly as f16 for the output sepconv.
__global__ __launch_bounds__(256) void k_dcnv3(const float* __restrict__ x,     // [TN,H,W,C]
                                               const float* __restrict__ offb,  // per-tn NCHW block, read flat
                                               const float* __restrict__ mkb,
                                               _Float16* __restrict__ y16) {
    const int wid = blockIdx.x * 8 + (threadIdx.x >> 5);
    const int lane = threadIdx.x & 31;
    const int g = wid & 3;
    const int hw = (wid >> 2) & 4095;
    const int tn = wid >> 14;
    if (tn >= TN_) return;
    const int h = hw >> 6, w = hw & 63;
    const int c = g * GC_ + lane;

    const float* offp = offb + (size_t)tn * 72 * HWDIM + (size_t)hw * 72 + g * (2 * PTAPS);
    const float* mkp  = mkb  + (size_t)tn * 36 * HWDIM + (size_t)hw * 36 + g * PTAPS;

    const float inv66 = 1.0f / 66.0f;
    const float refx = ((float)w + 1.5f) * inv66;
    const float refy = ((float)h + 1.5f) * inv66;

    float acc = 0.0f;
    #pragma unroll
    for (int p = 0; p < PTAPS; ++p) {
        const int kx = p / 3 - 1, ky = p % 3 - 1;   // x-major tap flatten (meshgrid 'ij')
        const float lx = refx + (float)kx * inv66 + offp[2 * p + 0] * inv66;
        const float ly = refy + (float)ky * inv66 + offp[2 * p + 1] * inv66;
        // sg = 2*loc-1; ix = ((sg_x+1)*Win - 1)/2 = loc_x*66 - 0.5 (align_corners=False)
        const float ix = lx * 66.0f - 0.5f;
        const float iy = ly * 66.0f - 0.5f;
        const float x0f = floorf(ix), y0f = floorf(iy);
        const float wx = ix - x0f, wy = iy - y0f;
        const int x0 = (int)x0f, y0 = (int)y0f;
        const float m = mkp[p];

        float v00 = 0.f, v01 = 0.f, v10 = 0.f, v11 = 0.f;
        // padded image is 66x66 with a ring of zeros; nonzero iff coord in [1,64]
        #define SAMP(xi, yi, dst)                                                      \
            if ((xi) >= 1 && (xi) <= 64 && (yi) >= 1 && (yi) <= 64)                    \
                dst = x[(((size_t)tn * HDIM + ((yi) - 1)) * WDIM + ((xi) - 1)) * CDIM + c];
        SAMP(x0,     y0,     v00)
        SAMP(x0 + 1, y0,     v01)
        SAMP(x0,     y0 + 1, v10)
        SAMP(x0 + 1, y0 + 1, v11)
        #undef SAMP

        acc += m * ((1.0f - wy) * ((1.0f - wx) * v00 + wx * v01) +
                    wy * ((1.0f - wx) * v10 + wx * v11));
    }
    y16[((size_t)tn * HWDIM + hw) * CDIM + c] = (_Float16)spikef(acc);
}

// ---------------- host-side launcher ----------------
extern "C" void kernel_launch(void* const* d_in, const int* in_sizes, int n_in,
                              void* d_out, int out_size, void* d_ws, size_t ws_size,
                              hipStream_t stream) {
    (void)in_sizes; (void)n_in; (void)out_size; (void)ws_size;

    const float* input    = (const float*)d_in[0];
    const float* dw_w     = (const float*)d_in[1];
    const float* dw_bn_s  = (const float*)d_in[2];
    const float* dw_bn_b  = (const float*)d_in[3];
    const float* off_w    = (const float*)d_in[4];
    const float* off_b    = (const float*)d_in[5];
    const float* off_bn_s = (const float*)d_in[6];
    const float* off_bn_b = (const float*)d_in[7];
    const float* mk_w     = (const float*)d_in[8];
    const float* mk_b     = (const float*)d_in[9];
    const float* mk_bn_s  = (const float*)d_in[10];
    const float* mk_bn_b  = (const float*)d_in[11];
    const float* ip_pw1_w = (const float*)d_in[12];
    const float* ip_bn1_s = (const float*)d_in[13];
    const float* ip_bn1_b = (const float*)d_in[14];
    const float* ip_dw_w  = (const float*)d_in[15];
    const float* ip_bn2_s = (const float*)d_in[16];
    const float* ip_bn2_b = (const float*)d_in[17];
    const float* ip_pw2_w = (const float*)d_in[18];
    const float* ip_bn3_s = (const float*)d_in[19];
    const float* ip_bn3_b = (const float*)d_in[20];
    const float* op_pw1_w = (const float*)d_in[21];
    const float* op_bn1_s = (const float*)d_in[22];
    const float* op_bn1_b = (const float*)d_in[23];
    const float* op_dw_w  = (const float*)d_in[24];
    const float* op_bn2_s = (const float*)d_in[25];
    const float* op_bn2_b = (const float*)d_in[26];
    const float* op_pw2_w = (const float*)d_in[27];
    const float* op_bn3_s = (const float*)d_in[28];
    const float* op_bn3_b = (const float*)d_in[29];

    // deterministic workspace carve
    char* base = (char*)d_ws;
    size_t o = 0;
    auto carve = [&](size_t bytes) { char* p = base + o; o += (bytes + 255) & ~(size_t)255; return p; };
    _Float16* act_a   = (_Float16*)carve((size_t)MROWS * CDIM * 2);   // spike(input) f16
    _Float16* med1    = (_Float16*)carve((size_t)MROWS * MEDDIM * 2); // pw1 out
    _Float16* med2    = (_Float16*)carve((size_t)MROWS * MEDDIM * 2); // dw out
    float*    x_f32   = (float*)   carve((size_t)MROWS * CDIM * 4);   // sepconv(in) result
    _Float16* x1_f16  = (_Float16*)carve((size_t)MROWS * CDIM * 2);   // branch dw out
    float*    offbuf  = (float*)   carve((size_t)MROWS * 72 * 4);
    float*    mkbuf   = (float*)   carve((size_t)MROWS * 36 * 4);
    _Float16* y_f16   = (_Float16*)carve((size_t)MROWS * CDIM * 2);   // spike(dcnv3 out)
    _Float16* wpw1P   = (_Float16*)carve((size_t)MEDDIM * CDIM * 2);  // [512][128]
    _Float16* wpw2P   = (_Float16*)carve((size_t)CDIM * MEDDIM * 2);  // [128][512]
    _Float16* woffP   = (_Float16*)carve((size_t)128 * CDIM * 2);     // [128][128] (72 padded)
    _Float16* wmkP    = (_Float16*)carve((size_t)128 * CDIM * 2);     // [128][128] (36 padded)
    _Float16* wopw1P  = (_Float16*)carve((size_t)MEDDIM * CDIM * 2);
    _Float16* wopw2P  = (_Float16*)carve((size_t)CDIM * MEDDIM * 2);

    const int nElemC = MROWS * CDIM;                 // 4,194,304
    const dim3 blk(256);

    // weight convert + pad into [Npad][K] fp16 (row = Cout, natural B-fragment layout)
    k_wpad<<<(MEDDIM * CDIM + 255) / 256, blk, 0, stream>>>(ip_pw1_w, wpw1P, MEDDIM, MEDDIM, CDIM);
    k_wpad<<<(CDIM * MEDDIM + 255) / 256, blk, 0, stream>>>(ip_pw2_w, wpw2P, CDIM, CDIM, MEDDIM);
    k_wpad<<<(128 * CDIM + 255) / 256,   blk, 0, stream>>>(off_w, woffP, 72, 128, CDIM);
    k_wpad<<<(128 * CDIM + 255) / 256,   blk, 0, stream>>>(mk_w, wmkP, 36, 128, CDIM);
    k_wpad<<<(MEDDIM * CDIM + 255) / 256, blk, 0, stream>>>(op_pw1_w, wopw1P, MEDDIM, MEDDIM, CDIM);
    k_wpad<<<(CDIM * MEDDIM + 255) / 256, blk, 0, stream>>>(op_pw2_w, wopw2P, CDIM, CDIM, MEDDIM);

    // 1) spike(input) -> f16 (feeds both input-sepconv pw1 and the offset branch)
    k_spike<<<(nElemC + 255) / 256, blk, 0, stream>>>(input, act_a, nElemC);

    // 2) input sepconv: pw1 (WMMA) + BN1 + spike -> f16 [M,512]
    k_gemm<1, CDIM><<<dim3(MROWS / 64, MEDDIM / 128), blk, 0, stream>>>(
        act_a, wpw1P, ip_bn1_s, ip_bn1_b, nullptr, nullptr, med1, MEDDIM);
    // 3) depthwise 3x3 + BN2 + spike
    k_dw<<<(MROWS * MEDDIM + 255) / 256, blk, 0, stream>>>(med1, ip_dw_w, ip_bn2_s, ip_bn2_b, med2, MEDDIM);
    // 4) pw2 (WMMA) + BN3 -> f32 value tensor x
    k_gemm<0, MEDDIM><<<dim3(MROWS / 64, CDIM / 128), blk, 0, stream>>>(
        med2, wpw2P, ip_bn3_s, ip_bn3_b, nullptr, x_f32, nullptr, CDIM);

    // 5) offset branch depthwise 3x3 + BN + spike on spike(input)
    k_dw<<<(nElemC + 255) / 256, blk, 0, stream>>>(act_a, dw_w, dw_bn_s, dw_bn_b, x1_f16, CDIM);
    // 6) offset 1x1 (WMMA) + bias + BN, faithful raw-reshape store
    k_gemm<2, CDIM><<<dim3(MROWS / 64, 1), blk, 0, stream>>>(
        x1_f16, woffP, off_bn_s, off_bn_b, off_b, offbuf, nullptr, 72);
    // 7) mask 1x1 (WMMA) + bias + BN + spike, faithful raw-reshape store
    k_gemm<3, CDIM><<<dim3(MROWS / 64, 1), blk, 0, stream>>>(
        x1_f16, wmkP, mk_bn_s, mk_bn_b, mk_b, mkbuf, nullptr, 36);

    // 8) deformable sampling core, fused spike -> f16
    k_dcnv3<<<(TN_ * HWDIM * GRPS) / 8, blk, 0, stream>>>(x_f32, offbuf, mkbuf, y_f16);

    // 9) output sepconv
    k_gemm<1, CDIM><<<dim3(MROWS / 64, MEDDIM / 128), blk, 0, stream>>>(
        y_f16, wopw1P, op_bn1_s, op_bn1_b, nullptr, nullptr, med1, MEDDIM);
    k_dw<<<(MROWS * MEDDIM + 255) / 256, blk, 0, stream>>>(med1, op_dw_w, op_bn2_s, op_bn2_b, med2, MEDDIM);
    k_gemm<0, MEDDIM><<<dim3(MROWS / 64, CDIM / 128), blk, 0, stream>>>(
        med2, wopw2P, op_bn3_s, op_bn3_b, nullptr, (float*)d_out, nullptr, CDIM);
}
